// Unet_51505247813783
// MI455X (gfx1250) — compile-verified, matrix operand order checked by reading, not run on hardware
//
#include <hip/hip_runtime.h>

typedef __attribute__((ext_vector_type(16))) _Float16 v16h;
typedef __attribute__((ext_vector_type(8)))  _Float16 v8h;
typedef __attribute__((ext_vector_type(4)))  _Float16 v4h;
typedef __attribute__((ext_vector_type(4)))  float    v4f;
typedef __attribute__((ext_vector_type(8)))  float    v8f;

#define EPSV   1e-5f
#define SLOPE  0.2f

// =====================================================================
// Weight pre-pass: W (f32, row-major [Ktot, Nact]) -> Wt (f16, [Npad, Kpad]
// "column-major" = transposed, zero padded). One B fragment then becomes
// two contiguous global_load_b128 from Wt.
// =====================================================================
__global__ void k_wt(const float* __restrict__ W, _Float16* __restrict__ Wt,
                     int Ktot, int Nact, int Kpad, long total)
{
    long stride = (long)gridDim.x * blockDim.x;
    for (long i = (long)blockIdx.x * blockDim.x + threadIdx.x; i < total; i += stride) {
        int n = (int)(i / Kpad);
        int k = (int)(i % Kpad);
        float v = (n < Nact && k < Ktot) ? W[(size_t)k * Nact + n] : 0.f;
        Wt[i] = (_Float16)v;
    }
}

// =====================================================================
// Gathered / dense GEMM on WMMA f32_16x16x32_f16 (wave32).
//   out[m][n] = sum_k A(m,k) * W[k][n] + bias[n]
//   gather (nb != 0): A(m,k) = x[ nb[m*7 + k/Cin]*Cin + k%Cin ], Ktot = 7*Cin
//   dense  (nb == 0): A(m,k) = x[ m*Cin + k ],                   Ktot = Cin
//
// Block = 128 threads = 4 waves sharing one LDS-staged gathered A tile
// (fragment-permuted: lane's v16h is contiguous -> 2x ds_load_b128).
// A staged with float4 global loads + half4 ds_store_b64 (all hot GEMMs
// have Cin % 4 == 0; scalar fallback only for Cin==3). B fragments come
// straight from pre-transposed f16 Wt: 2x global_load_b128, no guards.
// K-step 64 => two v_wmma per barrier.
// =====================================================================
__global__ __launch_bounds__(128)
void k_gemm_wmma(const float* __restrict__ x, const int* __restrict__ nb,
                 const _Float16* __restrict__ Wt, const float* __restrict__ bias,
                 float* __restrict__ out, int M, int Nact, int Ktot, int Kpad,
                 int Cin, int cshift, int cmask)
{
    // Two K-chunks of a 16x32 A tile, fragment-permuted: [chunk][lane][16]
    __shared__ alignas(32) _Float16 sA[2 * 32 * 16];

    const int tid  = threadIdx.x;
    const int lane = tid & 31;
    const int wv   = tid >> 5;
    const int m0   = blockIdx.x * 16;
    const int n0   = (blockIdx.y * 4 + wv) * 16;
    const int mn   = lane & 15;     // A-row / B-col within tile
    const int hh   = lane >> 4;     // half-wave selector
    const int col  = n0 + mn;       // always < Npad (Wt padded)
    const bool vec4 = (nb == nullptr) || (cshift >= 0);

    const _Float16* wcol = Wt + (size_t)col * Kpad;

    v8f acc = {};

    for (int kb = 0; kb < Ktot; kb += 64) {
        if (kb + 64 < Kpad)
            __builtin_prefetch(wcol + kb + 64, 0, 0);

        // ---- stage A (16 rows x 64 K) into fragment-permuted LDS ----
        // 256 float4-groups / 128 threads = 2 per thread.
        #pragma unroll
        for (int i = 0; i < 2; ++i) {
            int g   = i * 128 + tid;        // 0..255
            int m   = g >> 4;               // row 0..15
            int k4  = (g & 15) << 2;        // 0,4,...,60
            int kk  = kb + k4;
            int row = m0 + m;
            v4h hv = {};
            if (row < M) {
                if (vec4) {
                    if (kk < Ktot) {        // Ktot % 4 == 0 on this path
                        int base;
                        if (nb) {
                            int j  = kk >> cshift;
                            int cc = kk & cmask;
                            base = nb[row * 7 + j] * Cin + cc;
                        } else {
                            base = row * Cin + kk;
                        }
                        v4f f = *reinterpret_cast<const v4f*>(x + base);
                        hv[0] = (_Float16)f[0]; hv[1] = (_Float16)f[1];
                        hv[2] = (_Float16)f[2]; hv[3] = (_Float16)f[3];
                    }
                } else {                    // Cin == 3 (first conv) only
                    #pragma unroll
                    for (int e = 0; e < 4; ++e) {
                        int kk2 = kk + e;
                        float f = 0.f;
                        if (kk2 < Ktot) {
                            int j  = kk2 / Cin;
                            int cc = kk2 - j * Cin;
                            f = x[nb[row * 7 + j] * Cin + cc];
                        }
                        hv[e] = (_Float16)f;
                    }
                }
            }
            // permuted destination: 4 slots are contiguous & 8B aligned
            int chunk = k4 >> 5;
            int kl    = k4 & 31;
            int owner = m + (((kl >> 3) & 1) << 4);
            int e0    = ((kl >> 4) << 3) | (((kl >> 2) & 1) << 2);
            *reinterpret_cast<v4h*>(&sA[(chunk << 9) + (owner << 4) + e0]) = hv;
        }
        __syncthreads();

        // ---- two chunks: A = 2x ds_load_b128, B = 2x global_load_b128 ----
        #pragma unroll
        for (int c = 0; c < 2; ++c) {
            v16h a = *reinterpret_cast<const v16h*>(&sA[(c << 9) + (lane << 4)]);
            const _Float16* wp = wcol + kb + (c << 5) + (hh << 3);
            v8h blo = *reinterpret_cast<const v8h*>(wp);
            v8h bhi = *reinterpret_cast<const v8h*>(wp + 16);
            v16h b  = __builtin_shufflevector(blo, bhi,
                          0, 1, 2, 3, 4, 5, 6, 7, 8, 9, 10, 11, 12, 13, 14, 15);
            acc = __builtin_amdgcn_wmma_f32_16x16x32_f16(
                      false, a, false, b, (short)0, acc, false, false);
        }
        __syncthreads();
    }

    // ---- store: lane = col, VGPR v -> row (v + 8*half) ----
    if (col < Nact) {
        float bv = bias ? bias[col] : 0.f;
        #pragma unroll
        for (int v = 0; v < 8; ++v) {
            int row = m0 + v + 8 * hh;
            if (row < M) out[(size_t)row * Nact + col] = acc[v] + bv;
        }
    }
}

// ---------------- batch-norm statistics (one channel per block) ----------
__global__ __launch_bounds__(256)
void k_stats(const float* __restrict__ y, float* __restrict__ mean,
             float* __restrict__ var, int M, int C)
{
    __shared__ float ss[256], sq[256];
    int c = blockIdx.x;
    float s = 0.f, q = 0.f;
    for (int r = threadIdx.x; r < M; r += 256) {
        float v = y[(size_t)r * C + c];
        s += v; q += v * v;
    }
    ss[threadIdx.x] = s; sq[threadIdx.x] = q;
    __syncthreads();
    for (int st = 128; st > 0; st >>= 1) {
        if (threadIdx.x < st) {
            ss[threadIdx.x] += ss[threadIdx.x + st];
            sq[threadIdx.x] += sq[threadIdx.x + st];
        }
        __syncthreads();
    }
    if (threadIdx.x == 0) {
        float m = ss[0] / (float)M;
        mean[c] = m;
        var[c]  = sq[0] / (float)M - m * m;   // biased var, matches x.var(0)
    }
}

// ---------------- BN apply + leaky relu ----------------
__global__ void k_bn_lrelu(const float* __restrict__ y,
                           const float* __restrict__ mean,
                           const float* __restrict__ var,
                           const float* __restrict__ g,
                           const float* __restrict__ b,
                           float* __restrict__ out, long total, int C)
{
    long stride = (long)gridDim.x * blockDim.x;
    for (long i = (long)blockIdx.x * blockDim.x + threadIdx.x; i < total; i += stride) {
        int c = (int)(i % C);
        float v = (y[i] - mean[c]) * rsqrtf(var[c] + EPSV) * g[c] + b[c];
        out[i] = v > 0.f ? v : SLOPE * v;
    }
}

// ---------------- 7-neighbor mean pool ----------------
__global__ void k_pool(const float* __restrict__ x, const int* __restrict__ nbf,
                       float* __restrict__ out, long total, int C)
{
    long stride = (long)gridDim.x * blockDim.x;
    for (long i = (long)blockIdx.x * blockDim.x + threadIdx.x; i < total; i += stride) {
        long v = i / C; int c = (int)(i % C);
        float s = 0.f;
        #pragma unroll
        for (int j = 0; j < 7; ++j) s += x[(size_t)nbf[v * 7 + j] * C + c];
        out[i] = s * (1.f / 7.f);
    }
}

// ---------------- upconv "top" gather: dst[t][c] = y[top[t]][c] ----------
__global__ void k_up_top(const float* __restrict__ y, const int* __restrict__ top,
                         float* __restrict__ dst, long total, int co)
{
    long stride = (long)gridDim.x * blockDim.x;
    for (long i = (long)blockIdx.x * blockDim.x + threadIdx.x; i < total; i += stride) {
        long t = i / co; int c = (int)(i % co);
        dst[(size_t)t * 2 * co + c] = y[(size_t)top[t] * co + c];
    }
}

// -------- upconv "down" pair-mean, torch .view(-1, co, 2).mean(2) layout ----
__global__ void k_up_down(const float* __restrict__ y, const int* __restrict__ dn,
                          float* __restrict__ dst, long total, int raw, int co)
{
    long stride = (long)gridDim.x * blockDim.x;
    for (long i = (long)blockIdx.x * blockDim.x + threadIdx.x; i < total; i += stride) {
        long t = i / co; int c = (int)(i % co);
        long g0 = t * 2 * (long)co + (long)c * 2;
        long g1 = g0 + 1;
        float v0 = y[(size_t)dn[g0 / co] * co + (int)(g0 % co)];
        float v1 = y[(size_t)dn[g1 / co] * co + (int)(g1 % co)];
        dst[(size_t)(raw + t) * 2 * co + c] = 0.5f * (v0 + v1);
    }
}

// ---------------- channel-concat copy into second half -------------------
__global__ void k_copy_cols(const float* __restrict__ src, float* __restrict__ dst,
                            long total, int co)
{
    long stride = (long)gridDim.x * blockDim.x;
    for (long i = (long)blockIdx.x * blockDim.x + threadIdx.x; i < total; i += stride) {
        long r = i / co; int c = (int)(i % co);
        dst[(size_t)r * 2 * co + co + c] = src[i];
    }
}

// =====================================================================
extern "C" void kernel_launch(void* const* d_in, const int* in_sizes, int n_in,
                              void* d_out, int out_size, void* d_ws, size_t ws_size,
                              hipStream_t stream)
{
    (void)in_sizes; (void)n_in; (void)out_size; (void)ws_size;
    static const int NL[5] = {40962, 10242, 2562, 642, 162};
    static const int CH[6] = {3, 64, 128, 256, 512, 1024};

    const float* X = (const float*)d_in[0];
    // JAX pytree flatten order (dict keys sorted): params.down -> params.out -> params.up
    // down[l]: bn1.b bn1.g bn2.b bn2.g conv1.W conv1.b conv2.W conv2.b
    auto DN  = [&](int l, int f) { return (const float*)d_in[1 + l * 8 + f]; };
    const float* OUT_W = (const float*)d_in[41];
    const float* OUT_B = (const float*)d_in[42];
    // up[j]: bn1.b bn1.g bn2.b bn2.g conv1.W conv1.b conv2.W conv2.b upconv.W upconv.b
    auto UP  = [&](int j, int f) { return (const float*)d_in[43 + j * 10 + f]; };
    auto NB  = [&](int l)  { return (const int*)d_in[83 + l]; };
    auto TOP = [&](int lt) { return (const int*)d_in[88 + lt]; };
    auto DWN = [&](int lt) { return (const int*)d_in[92 + lt]; };

    // ---- workspace bump allocator (floats) ----
    float* ws = (float*)d_ws;
    size_t off = 0;
    float* skip[5];
    for (int l = 0; l < 5; ++l) { skip[l] = ws + off; off += (size_t)NL[l] * CH[l + 1]; }
    float* pA = ws + off; off += (size_t)NL[0] * 2 * CH[1];   // 40962*128
    float* pB = ws + off; off += (size_t)NL[0] * 2 * CH[1];
    float* pC = ws + off; off += (size_t)NL[1] * 7 * CH[1];   // 10242*448
    float* pm = ws + off; off += 1024;
    float* pv = ws + off; off += 1024;
    _Float16* pWt = (_Float16*)(ws + off);                    // max 7168*1024 halfs
    off += (size_t)7168 * 1024 / 2 + 64;

    auto gemm = [&](const float* x, const int* nb, const float* W, const float* b,
                    float* o, int M, int Nact, int Ktot, int Cin) {
        int cshift = ((Cin & (Cin - 1)) == 0) ? __builtin_ctz(Cin) : -1;
        int cmask  = Cin - 1;
        int Kpad   = ((Ktot + 63) / 64) * 64;
        int Npad   = ((Nact + 63) / 64) * 64;
        long wtot  = (long)Npad * Kpad;
        k_wt<<<dim3((unsigned)((wtot + 255) / 256)), dim3(256), 0, stream>>>(
            W, pWt, Ktot, Nact, Kpad, wtot);
        dim3 grid((M + 15) / 16, (Nact + 63) / 64);
        k_gemm_wmma<<<grid, dim3(128), 0, stream>>>(x, nb, pWt, b, o, M, Nact,
                                                    Ktot, Kpad, Cin, cshift, cmask);
    };
    auto bnlr = [&](const float* y, const float* g, const float* b, float* o,
                    int M, int C) {
        k_stats<<<dim3(C), dim3(256), 0, stream>>>(y, pm, pv, M, C);
        long tot = (long)M * C;
        k_bn_lrelu<<<dim3((unsigned)((tot + 255) / 256)), dim3(256), 0, stream>>>(
            y, pm, pv, g, b, o, tot, C);
    };

    // -------------------- encoder --------------------
    for (int l = 0; l < 5; ++l) {
        int n = NL[l], co = CH[l + 1];
        const float* src; int ci;
        if (l == 0) { src = X; ci = 3; }
        else {
            long tot = (long)n * CH[l];
            k_pool<<<dim3((unsigned)((tot + 255) / 256)), dim3(256), 0, stream>>>(
                skip[l - 1], NB(l - 1), pA, tot, CH[l]);
            src = pA; ci = CH[l];
        }
        gemm(src, NB(l), DN(l, 4), DN(l, 5), pB, n, co, 7 * ci, ci);
        bnlr(pB, DN(l, 1), DN(l, 0), pA, n, co);
        gemm(pA, NB(l), DN(l, 6), DN(l, 7), pB, n, co, 7 * co, co);
        bnlr(pB, DN(l, 3), DN(l, 2), skip[l], n, co);
    }

    // -------------------- decoder --------------------
    float* h = skip[4];
    for (int j = 0; j < 4; ++j) {
        int lt  = 3 - j;
        int co  = CH[lt + 1], ci = CH[5 - j];
        int raw = NL[lt + 1], nn = NL[lt];

        // upconv: dense GEMM [raw, ci] @ [ci, 7co] -> pC (== y as [raw*7, co])
        gemm(h, nullptr, UP(j, 8), UP(j, 9), pC, raw, 7 * co, ci, ci);

        float* cat = (h == pA) ? pB : pA;
        long t1 = (long)raw * co;
        k_up_top<<<dim3((unsigned)((t1 + 255) / 256)), dim3(256), 0, stream>>>(
            pC, TOP(lt), cat, t1, co);
        long t2 = (long)(nn - raw) * co;
        k_up_down<<<dim3((unsigned)((t2 + 255) / 256)), dim3(256), 0, stream>>>(
            pC, DWN(lt), cat, t2, raw, co);
        long t3 = (long)nn * co;
        k_copy_cols<<<dim3((unsigned)((t3 + 255) / 256)), dim3(256), 0, stream>>>(
            skip[lt], cat, t3, co);

        float* oth = (cat == pA) ? pB : pA;
        gemm(cat, NB(lt), UP(j, 4), UP(j, 5), oth, nn, co, 14 * co, 2 * co);
        bnlr(oth, UP(j, 1), UP(j, 0), cat, nn, co);
        gemm(cat, NB(lt), UP(j, 6), UP(j, 7), oth, nn, co, 7 * co, co);
        bnlr(oth, UP(j, 3), UP(j, 2), cat, nn, co);
        h = cat;
    }

    // -------------------- output head: [40962,64] @ [64,36] --------------------
    gemm(h, nullptr, OUT_W, OUT_B, (float*)d_out, NL[0], 36, CH[1], CH[1]);
}
